// BSTGCNet_25469156065429
// MI455X (gfx1250) — matrix operand order repaired and checked by reading, not compiled, not run on hardware
//
#include <hip/hip_runtime.h>
#include <hip/hip_bf16.h>
#include <math.h>

typedef __attribute__((ext_vector_type(16))) _Float16 v16h;
typedef __attribute__((ext_vector_type(8)))  float    v8f;
typedef __attribute__((ext_vector_type(4)))  int      v4i;

#define NNODE 512
#define HDIM  64
#define BT    96
#define BB    8
#define TT    12
#define FINC  2
#define PP    12
#define KCH   128          // K-chunk staged in LDS
#define KPAD  8            // row pad (8 f16 = 16B, keeps B128 segments aligned)

#define AS1 __attribute__((address_space(1)))
#define AS3 __attribute__((address_space(3)))

#if __has_builtin(__builtin_amdgcn_global_load_async_to_lds_b128) && \
    __has_builtin(__builtin_amdgcn_s_wait_asynccnt)
#define HAS_ASYNC_LDS 1
#else
#define HAS_ASYNC_LDS 0
#endif

#if HAS_ASYNC_LDS
#define WAIT_ASYNC(n) __builtin_amdgcn_s_wait_asynccnt(n)
#else
#define WAIT_ASYNC(n)
#endif

__device__ __forceinline__ float wave_reduce_sum(float v) {
#pragma unroll
  for (int off = 16; off >= 1; off >>= 1) v += __shfl_xor(v, off, 32);
  return v;
}
__device__ __forceinline__ float wave_reduce_max(float v) {
#pragma unroll
  for (int off = 16; off >= 1; off >>= 1) v = fmaxf(v, __shfl_xor(v, off, 32));
  return v;
}

// ---------------------------------------------------------------------------
// Kernel 1: Wh = X @ W (K=2); store transposed f16 Whh[m][h][k-node] for the
// WMMA B operand, plus s1 = Wh@a1, s2 = Wh@a2 per node.
// grid (N, BT), block HDIM (= 2 waves)
// ---------------------------------------------------------------------------
__global__ void wh_s1s2_kernel(const float* __restrict__ X, const float* __restrict__ W,
                               const float* __restrict__ a1, const float* __restrict__ a2,
                               _Float16* __restrict__ Whh, float* __restrict__ s1,
                               float* __restrict__ s2) {
  const int n = blockIdx.x, m = blockIdx.y, h = threadIdx.x;
  const size_t node = (size_t)m * NNODE + n;
  const float x0 = X[node * FINC + 0];
  const float x1 = X[node * FINC + 1];
  const float wh = fmaf(x0, W[h], x1 * W[HDIM + h]);          // W is [2, H]
  Whh[((size_t)m * HDIM + h) * NNODE + n] = (_Float16)wh;     // transposed, f16

  float p1 = wave_reduce_sum(wh * a1[h]);
  float p2 = wave_reduce_sum(wh * a2[h]);
  __shared__ float red[4];
  const int wv = h >> 5;
  if ((h & 31) == 0) { red[wv] = p1; red[2 + wv] = p2; }
  __syncthreads();
  if (h == 0) {
    s1[node] = red[0] + red[1];
    s2[node] = red[2] + red[3];
  }
}

// ---------------------------------------------------------------------------
// Async (or fallback) staging of one 64 x 128 f16 Wh chunk into LDS.
// 64 rows x 256B = 1024 x 16B transfers; 128 threads -> 8 B128 ops each.
// ---------------------------------------------------------------------------
__device__ __forceinline__ void issue_chunk(const _Float16* __restrict__ whg,
                                            _Float16 (*buf)[KCH + KPAD],
                                            int chunk, int tid) {
#if HAS_ASYNC_LDS
#pragma unroll
  for (int i = 0; i < 8; ++i) {
    const int q = tid + 128 * i;
    const int row = q >> 4;        // 0..63
    const int seg = q & 15;        // 16B segment within the 256B row
    const _Float16* g = whg + (size_t)row * NNODE + chunk * KCH + seg * 8;
    __builtin_amdgcn_global_load_async_to_lds_b128(
        (AS1 v4i*)(AS1 void*)g,
        (AS3 v4i*)(AS3 void*)&buf[row][seg * 8], 0, 0);
  }
#else
  for (int idx = tid; idx < HDIM * KCH; idx += 128) {
    const int row = idx >> 7, kk = idx & (KCH - 1);
    buf[row][kk] = whg[(size_t)row * NNODE + chunk * KCH + kk];
  }
#endif
}

// ---------------------------------------------------------------------------
// One K-chunk of the att @ Wh WMMA accumulation (4 x v_wmma per call).
// ---------------------------------------------------------------------------
__device__ __forceinline__ v8f compute_chunk(v8f acc,
                                             const _Float16 (*whb)[KCH + KPAD],
                                             const _Float16 (*att)[NNODE + 16],
                                             int chunk, int half, int r, int ncol) {
#pragma unroll
  for (int k0 = 0; k0 < KCH; k0 += 32) {
    v16h a, b;
    // A fragment: 16x32 f16 (ISA 7.12.2 layout; lane half selects K sub-block)
#pragma unroll
    for (int v = 0; v < 8; ++v) {
      const int ka = (v < 4) ? (8 * half + 2 * v) : (16 + 8 * half + 2 * (v - 4));
      a[2 * v]     = att[r][chunk * KCH + k0 + ka];
      a[2 * v + 1] = att[r][chunk * KCH + k0 + ka + 1];
    }
    // B fragment: 32x16 f16; same K order as A (e<8: K=8*half+e, e>=8: K=16+8*half+e-8)
#pragma unroll
    for (int e = 0; e < 16; ++e)
      b[e] = whb[ncol][k0 + ((e < 8) ? (8 * half + e) : (16 + 8 * half + (e - 8)))];
    acc = __builtin_amdgcn_wmma_f32_16x16x32_f16(
        false, a, false, b, (short)0, acc, false, false);
  }
  return acc;
}

// ---------------------------------------------------------------------------
// Kernel 2: GAT attention.  Per (m, 16-row tile):
//  - issue async DMA of Wh chunks 0,1 (overlaps the whole softmax phase)
//  - masked softmax into f16 LDS tile
//  - double-buffered WMMA accumulation over K=512
// grid (N/16, BT), block 128 (= 4 waves, one 16x16 n-tile each)
// ---------------------------------------------------------------------------
__global__ __launch_bounds__(128)
void gat_attn_kernel(const _Float16* __restrict__ Whh, const float* __restrict__ s1,
                     const float* __restrict__ s2, const int* __restrict__ adj,
                     float* __restrict__ out, const int accumulate) {
  __shared__ _Float16 att_s[16][NNODE + 16];                   // softmaxed rows (f16)
  __shared__ __align__(16) _Float16 wh_c[2][HDIM][KCH + KPAD]; // double-buffered Wh

  const int tid  = threadIdx.x;
  const int wave = tid >> 5, lane = tid & 31;
  const int tile = blockIdx.x, m = blockIdx.y;
  const int row0 = tile * 16;
  const size_t mbase = (size_t)m * NNODE;
  const _Float16* whg = Whh + (size_t)m * HDIM * NNODE;        // [64][512] f16 tile

  __builtin_prefetch(s2 + mbase, 0, 1);

  // Kick off DMA for the first two chunks; lands while we do softmax.
  issue_chunk(whg, wh_c[0], 0, tid);
  issue_chunk(whg, wh_c[1], 1, tid);

  // ---- Phase 1: masked softmax, 4 rows per wave, 16 cols per lane ----
  for (int rr = 0; rr < 4; ++rr) {
    const int i = wave * 4 + rr;
    const int grow = row0 + i;
    const float s1v = s1[mbase + grow];
    float ev[16];
    float mx = -3.0e38f;
#pragma unroll
    for (int jj = 0; jj < 16; ++jj) {
      const int j = lane + 32 * jj;
      float e = s1v + s2[mbase + j];
      e = (e > 0.f) ? e : 0.2f * e;                            // leaky_relu(0.2)
      e = (adj[(size_t)grow * NNODE + j] > 0) ? e : -9.0e15f;  // mask
      ev[jj] = e;
      mx = fmaxf(mx, e);
    }
    mx = wave_reduce_max(mx);
    float sum = 0.f;
#pragma unroll
    for (int jj = 0; jj < 16; ++jj) { ev[jj] = __expf(ev[jj] - mx); sum += ev[jj]; }
    sum = wave_reduce_sum(sum);
    const float inv = 1.f / sum;
#pragma unroll
    for (int jj = 0; jj < 16; ++jj)
      att_s[i][lane + 32 * jj] = (_Float16)(ev[jj] * inv);
  }

  // ---- Phase 2: double-buffered WMMA pipeline over K = 4 x 128 ----
  const int half = lane >> 4, r = lane & 15;
  const int ncol = wave * 16 + r;
  v8f acc = {};

  WAIT_ASYNC(8);                 // chunk 0 complete (async done in order)
  __syncthreads();               // + softmax tile visible to all waves
  acc = compute_chunk(acc, wh_c[0], att_s, 0, half, r, ncol);
  __syncthreads();

  issue_chunk(whg, wh_c[0], 2, tid);
  WAIT_ASYNC(8);                 // chunk 1 complete
  __syncthreads();
  acc = compute_chunk(acc, wh_c[1], att_s, 1, half, r, ncol);
  __syncthreads();

  issue_chunk(whg, wh_c[1], 3, tid);
  WAIT_ASYNC(8);                 // chunk 2 complete
  __syncthreads();
  acc = compute_chunk(acc, wh_c[0], att_s, 2, half, r, ncol);
  __syncthreads();

  WAIT_ASYNC(0);                 // chunk 3 complete
  __syncthreads();
  acc = compute_chunk(acc, wh_c[1], att_s, 3, half, r, ncol);

  // ---- Phase 3: ELU + writeback (D layout: VGPR v -> row v + 8*half) ----
#pragma unroll
  for (int v = 0; v < 8; ++v) {
    const int grow = row0 + v + 8 * half;
    const float d = acc[v];
    const float ev = (d > 0.f) ? d : expm1f(d);   // elu
    float* p = &out[(mbase + grow) * HDIM + ncol];
    if (accumulate) *p += ev; else *p = ev;
  }
}

// ---------------------------------------------------------------------------
// Kernel 3: spatial = relu(concat(f_s, f_nd) @ Wf + bf)
// grid (N, BT), block HDIM
// ---------------------------------------------------------------------------
__global__ void fuse_kernel(const float* __restrict__ f_s, const float* __restrict__ f_nd,
                            const float* __restrict__ Wf, const float* __restrict__ bf,
                            float* __restrict__ spatial) {
  __shared__ float fin[2 * HDIM];
  const int n = blockIdx.x, m = blockIdx.y, h = threadIdx.x;
  const size_t node = (size_t)m * NNODE + n;
  fin[h]        = f_s[node * HDIM + h];
  fin[HDIM + h] = f_nd[node * HDIM + h];
  __syncthreads();
  float acc = bf[h];
#pragma unroll 8
  for (int k = 0; k < 2 * HDIM; ++k) acc = fmaf(fin[k], Wf[k * HDIM + h], acc);
  spatial[node * HDIM + h] = fmaxf(acc, 0.f);
}

// ---------------------------------------------------------------------------
// Kernel 4: GRU over T=12; one block per (b,n) row, h state in LDS.
// grid B*N, block HDIM
// ---------------------------------------------------------------------------
__global__ void gru_kernel(const float* __restrict__ spatial, const float* __restrict__ X,
                           const float* __restrict__ W_ih, const float* __restrict__ W_hh,
                           const float* __restrict__ b_ih, const float* __restrict__ b_hh,
                           float* __restrict__ h_last) {
  __shared__ float inb[HDIM + FINC];
  __shared__ float hb[HDIM];
  __shared__ float hn[HDIM];
  const int bn = blockIdx.x;
  const int b = bn >> 9, n = bn & (NNODE - 1);
  const int j = threadIdx.x;
  const int K = HDIM + FINC;
  hb[j] = 0.f;
  for (int t = 0; t < TT; ++t) {
    __syncthreads();
    const size_t node = ((size_t)(b * TT + t)) * NNODE + n;
    inb[j] = spatial[node * HDIM + j];
    if (j < FINC) inb[HDIM + j] = X[node * FINC + j];
    __syncthreads();
    float gxr = b_ih[j], gxz = b_ih[HDIM + j], gxn = b_ih[2 * HDIM + j];
    for (int k = 0; k < K; ++k) {
      const float xv = inb[k];
      gxr = fmaf(xv, W_ih[(size_t)j * K + k], gxr);
      gxz = fmaf(xv, W_ih[(size_t)(HDIM + j) * K + k], gxz);
      gxn = fmaf(xv, W_ih[(size_t)(2 * HDIM + j) * K + k], gxn);
    }
    float ghr = b_hh[j], ghz = b_hh[HDIM + j], ghn = b_hh[2 * HDIM + j];
    for (int k = 0; k < HDIM; ++k) {
      const float hv = hb[k];
      ghr = fmaf(hv, W_hh[(size_t)j * HDIM + k], ghr);
      ghz = fmaf(hv, W_hh[(size_t)(HDIM + j) * HDIM + k], ghz);
      ghn = fmaf(hv, W_hh[(size_t)(2 * HDIM + j) * HDIM + k], ghn);
    }
    const float rg = 1.f / (1.f + __expf(-(gxr + ghr)));
    const float zg = 1.f / (1.f + __expf(-(gxz + ghz)));
    const float ng = tanhf(gxn + rg * ghn);
    hn[j] = (1.f - zg) * ng + zg * hb[j];
    __syncthreads();
    hb[j] = hn[j];
  }
  h_last[(size_t)bn * HDIM + j] = hb[j];
}

// ---------------------------------------------------------------------------
// Kernel 5: pred = relu(h @ W1 + b1) @ W2 + b2, output [B, P, N]
// grid B*N, block 32
// ---------------------------------------------------------------------------
__global__ void head_kernel(const float* __restrict__ h_last, const float* __restrict__ W1,
                            const float* __restrict__ b1, const float* __restrict__ W2,
                            const float* __restrict__ b2, float* __restrict__ out) {
  __shared__ float hid[32];
  const int bn = blockIdx.x, j = threadIdx.x;
  const int b = bn >> 9, n = bn & (NNODE - 1);
  float acc = b1[j];
  for (int k = 0; k < HDIM; ++k)
    acc = fmaf(h_last[(size_t)bn * HDIM + k], W1[k * 32 + j], acc);
  hid[j] = fmaxf(acc, 0.f);
  __syncthreads();
  if (j < PP) {
    float o = b2[j];
    for (int k = 0; k < 32; ++k) o = fmaf(hid[k], W2[k * PP + j], o);
    out[(size_t)b * PP * NNODE + (size_t)j * NNODE + n] = o;
  }
}

// ---------------------------------------------------------------------------
extern "C" void kernel_launch(void* const* d_in, const int* in_sizes, int n_in,
                              void* d_out, int out_size, void* d_ws, size_t ws_size,
                              hipStream_t stream) {
  const float* X    = (const float*)d_in[0];
  const int*   Gs   = (const int*)d_in[1];
  const int*   Gn   = (const int*)d_in[2];
  const int*   Gd   = (const int*)d_in[3];
  const float* Wg   = (const float*)d_in[4];
  const float* a1g  = (const float*)d_in[5];
  const float* a2g  = (const float*)d_in[6];
  const float* Wn   = (const float*)d_in[7];
  const float* a1n  = (const float*)d_in[8];
  const float* a2n  = (const float*)d_in[9];
  const float* Wd   = (const float*)d_in[10];
  const float* a1d  = (const float*)d_in[11];
  const float* a2d  = (const float*)d_in[12];
  const float* Wf   = (const float*)d_in[13];
  const float* bf   = (const float*)d_in[14];
  const float* W_ih = (const float*)d_in[15];
  const float* W_hh = (const float*)d_in[16];
  const float* b_ih = (const float*)d_in[17];
  const float* b_hh = (const float*)d_in[18];
  const float* W1   = (const float*)d_in[19];
  const float* b1   = (const float*)d_in[20];
  const float* W2   = (const float*)d_in[21];
  const float* b2   = (const float*)d_in[22];
  float* out = (float*)d_out;

  float* ws = (float*)d_ws;
  const size_t S = (size_t)BT * NNODE * HDIM;  // 3,145,728
  const size_t V = (size_t)BT * NNODE;         // 49,152
  size_t off = 0;
  float* f_s     = ws + off; off += S;
  float* f_nd    = ws + off; off += S;
  float* spatial = ws + off; off += S;
  float* h_last  = ws + off; off += (size_t)BB * NNODE * HDIM;
  float* s1s = ws + off; off += V;  float* s2s = ws + off; off += V;
  float* s1n = ws + off; off += V;  float* s2n = ws + off; off += V;
  float* s1d = ws + off; off += V;  float* s2d = ws + off; off += V;
  _Float16* whh = (_Float16*)(ws + off);
  _Float16* Whh_s = whh;
  _Float16* Whh_n = whh + S;
  _Float16* Whh_d = whh + 2 * S;

  dim3 gNode(NNODE, BT);
  wh_s1s2_kernel<<<gNode, HDIM, 0, stream>>>(X, Wg, a1g, a2g, Whh_s, s1s, s2s);
  wh_s1s2_kernel<<<gNode, HDIM, 0, stream>>>(X, Wn, a1n, a2n, Whh_n, s1n, s2n);
  wh_s1s2_kernel<<<gNode, HDIM, 0, stream>>>(X, Wd, a1d, a2d, Whh_d, s1d, s2d);

  dim3 gAtt(NNODE / 16, BT);
  gat_attn_kernel<<<gAtt, 128, 0, stream>>>(Whh_s, s1s, s2s, Gs, f_s, 0);
  gat_attn_kernel<<<gAtt, 128, 0, stream>>>(Whh_n, s1n, s2n, Gn, f_nd, 0);
  gat_attn_kernel<<<gAtt, 128, 0, stream>>>(Whh_d, s1d, s2d, Gd, f_nd, 1);

  fuse_kernel<<<gNode, HDIM, 0, stream>>>(f_s, f_nd, Wf, bf, spatial);
  gru_kernel<<<BB * NNODE, HDIM, 0, stream>>>(spatial, X, W_ih, W_hh, b_ih, b_hh, h_last);
  head_kernel<<<BB * NNODE, 32, 0, stream>>>(h_last, W1, b1, W2, b2, out);
}
